// FinancialReasoningGNN_55903294324884
// MI455X (gfx1250) — matrix operand order repaired and checked by reading, not compiled, not run on hardware
//
#include <hip/hip_runtime.h>
#include <hip/hip_bf16.h>

typedef _Float16 v16h __attribute__((ext_vector_type(16)));
typedef _Float16 h8   __attribute__((ext_vector_type(8)));
typedef float    v8f  __attribute__((ext_vector_type(8)));

__device__ __forceinline__ v16h cat16(h8 lo, h8 hi) {
  return __builtin_shufflevector(lo, hi, 0,1,2,3,4,5,6,7,8,9,10,11,12,13,14,15);
}
// A fragment (16x32 f16): lane l holds row m=l&15; halfs 0-7 = K kb+koff..+7,
// halfs 8-15 = K kb+16+koff..+7, koff = 8*(l>=16). Row must be k-contiguous f16.
__device__ __forceinline__ v16h ld_afrag(const _Float16* row, int kb, int koff) {
  h8 lo = *(const h8*)(row + kb + koff);
  h8 hi = *(const h8*)(row + kb + 16 + koff);
  return cat16(lo, hi);
}
// B fragment (32x16 f16): lane l holds col n=l&15; halfs 0-7 = K kb+kgrp/2..+7,
// halfs 8-15 = K kb+16+kgrp/2..+7, kgrp = 16*(l>=16). Row (n-major) must be k-contiguous f16.
__device__ __forceinline__ v16h ld_bfrag(const _Float16* row, int kb, int kgrp) {
  h8 lo = *(const h8*)(row + kb + (kgrp >> 1));
  h8 hi = *(const h8*)(row + kb + 16 + (kgrp >> 1));
  return cat16(lo, hi);
}
#define WMMA_F16(a, b, c) \
  __builtin_amdgcn_wmma_f32_16x16x32_f16(false, (a), false, (b), (short)0, (c), false, false)

__device__ __forceinline__ unsigned enc_f32(float f) {
  unsigned b = __float_as_uint(f);
  return (b & 0x80000000u) ? ~b : (b | 0x80000000u);
}
__device__ __forceinline__ float dec_f32(unsigned u) {
  unsigned b = (u & 0x80000000u) ? (u & 0x7fffffffu) : ~u;
  return __uint_as_float(b);
}
// tanh via hardware exp + rcp (co-issues with WMMA/VALU); clamp avoids inf*0.
__device__ __forceinline__ float fast_tanh(float t) {
  float tc = fminf(fmaxf(t, -9.f), 9.f);
  float ex = __expf(2.f * tc);
  return (ex - 1.f) * __builtin_amdgcn_rcpf(ex + 1.f);
}

// ---------------- init: zero accumulators, -inf segment-max ----------------
__global__ void gnn_init_kernel(float* agg, float* out2, float* asum,
                                unsigned* smax, int N) {
  long t = (long)blockIdx.x * blockDim.x + threadIdx.x;
  long n64 = (long)N * 64, n256 = (long)N * 256, n4 = (long)N * 4;
  if (t < n64)  agg[t]  = 0.f;
  if (t < n256) out2[t] = 0.f;
  if (t < n4) { asum[t] = 0.f; smax[t] = 0x007FFFFFu; /* enc(-inf) */ }
}

// ---------------- weight transpose+convert for fusion GEMMs ----------------
__global__ void gnn_convw_kernel(const float* wf1, const float* wf2,
                                 _Float16* wf1t, _Float16* wf2t) {
  int t = blockIdx.x * blockDim.x + threadIdx.x;
  if (t < 320 * 128) { int k = t / 128, n = t % 128; wf1t[n * 320 + k] = (_Float16)wf1[t]; }
  if (t < 128 * 32)  { int k = t / 32,  n = t % 32;  wf2t[n * 128 + k] = (_Float16)wf2[t]; }
}

// ---------------- NNConv edge pipeline (WMMA GEMM E x 64 x 192) ------------
__global__ void __launch_bounds__(256)
gnn_nnconv_kernel(const float* __restrict__ x, const int* __restrict__ ei,
                  const float* __restrict__ ea, const float* __restrict__ w1,
                  const float* __restrict__ b1, const float* __restrict__ w2,
                  const float* __restrict__ b2, const float* __restrict__ ln_g,
                  const float* __restrict__ ln_b, float* __restrict__ agg,
                  int N, int E) {
  __shared__ _Float16 w2t[192 * 64];              // [n][k], f16, 24 KB
  __shared__ float w1s[64], b1s[64];
  __shared__ float b2s[192], gs[192], bs[192];
  __shared__ float xsl[8][16][3];
  __shared__ int   dsl[8][16];

  const int tid = threadIdx.x;
  for (int t = tid; t < 192 * 64; t += 256) {
    int n = t >> 6, k = t & 63;
    w2t[t] = (_Float16)w2[k * 192 + n];
  }
  if (tid < 64)  { w1s[tid] = w1[tid]; b1s[tid] = b1[tid]; }
  if (tid < 192) { b2s[tid] = b2[tid]; gs[tid] = ln_g[tid]; bs[tid] = ln_b[tid]; }

  const int wave = tid >> 5, lane = tid & 31;
  const int ebase = (blockIdx.x * 8 + wave) * 16;
  if (lane < 16) {
    int e = ebase + lane;
    if (e < E) {
      int s = ei[e];
      xsl[wave][lane][0] = x[s * 3 + 0];
      xsl[wave][lane][1] = x[s * 3 + 1];
      xsl[wave][lane][2] = x[s * 3 + 2];
      dsl[wave][lane] = ei[E + e];
    } else {
      xsl[wave][lane][0] = 0.f; xsl[wave][lane][1] = 0.f; xsl[wave][lane][2] = 0.f;
      dsl[wave][lane] = -1;
    }
  }
  __syncthreads();

  const int col = lane & 15, hi = lane >> 4, koff = hi * 8, kgrp = hi * 16;
  const int e = ebase + col;
  const float eav = (e < E) ? ea[e] : 0.f;

  // A fragments in registers: h[k] = relu(ea*w1[k] + b1[k]), K = 0..63
  v16h a0, a1;
#pragma unroll
  for (int i = 0; i < 8; ++i) {
    int kA = koff + i, kB = 16 + koff + i;
    float h;
    h = eav * w1s[kA] + b1s[kA];           a0[i]     = (_Float16)(h > 0.f ? h : 0.f);
    h = eav * w1s[kB] + b1s[kB];           a0[i + 8] = (_Float16)(h > 0.f ? h : 0.f);
    h = eav * w1s[32 + kA] + b1s[32 + kA]; a1[i]     = (_Float16)(h > 0.f ? h : 0.f);
    h = eav * w1s[32 + kB] + b1s[32 + kB]; a1[i + 8] = (_Float16)(h > 0.f ? h : 0.f);
  }

  // 12 n-tiles x 2 k-steps = 24 WMMAs for the 16x192 tile
  v8f acc[12];
#pragma unroll
  for (int nt = 0; nt < 12; ++nt) {
    const _Float16* brow = &w2t[(nt * 16 + col) * 64];
    v8f c = {};
    v16h bf0 = ld_bfrag(brow, 0, kgrp);
    c = WMMA_F16(a0, bf0, c);
    v16h bf1 = ld_bfrag(brow, 32, kgrp);
    c = WMMA_F16(a1, bf1, c);
    acc[nt] = c;
  }

  // bias + LayerNorm stats; row m = hi*8+v spans (col,nt) across the half-wave
  float sm[8], sq[8];
#pragma unroll
  for (int v = 0; v < 8; ++v) { sm[v] = 0.f; sq[v] = 0.f; }
#pragma unroll
  for (int nt = 0; nt < 12; ++nt) {
    float bb = b2s[col + 16 * nt];
#pragma unroll
    for (int v = 0; v < 8; ++v) {
      float val = acc[nt][v] + bb;
      acc[nt][v] = val;
      sm[v] += val; sq[v] += val * val;
    }
  }
#pragma unroll
  for (int msk = 1; msk <= 8; msk <<= 1) {
#pragma unroll
    for (int v = 0; v < 8; ++v) {
      sm[v] += __shfl_xor(sm[v], msk, 32);
      sq[v] += __shfl_xor(sq[v], msk, 32);
    }
  }
  float mean[8], rstd[8];
#pragma unroll
  for (int v = 0; v < 8; ++v) {
    float mu = sm[v] * (1.f / 192.f);
    float var = sq[v] * (1.f / 192.f) - mu * mu;
    mean[v] = mu;
    rstd[v] = rsqrtf(var + 1e-5f);
  }

  // tanh(LN(h)) -> w_edge[i][o] (n = i*64+o); msg[o] = sum_i x[src][i]*w_edge
#pragma unroll
  for (int v = 0; v < 8; ++v) {
    int m = hi * 8 + v;
    float x0 = xsl[wave][m][0], x1 = xsl[wave][m][1], x2 = xsl[wave][m][2];
    float msg0 = 0.f, msg1 = 0.f, msg2 = 0.f, msg3 = 0.f;
#pragma unroll
    for (int nt = 0; nt < 12; ++nt) {
      int n = col + 16 * nt;
      float t = (acc[nt][v] - mean[v]) * rstd[v] * gs[n] + bs[n];
      float w = fast_tanh(t);
      float xi = (nt < 4) ? x0 : (nt < 8 ? x1 : x2);
      float cb = w * xi;
      if ((nt & 3) == 0) msg0 += cb;
      else if ((nt & 3) == 1) msg1 += cb;
      else if ((nt & 3) == 2) msg2 += cb;
      else msg3 += cb;
    }
    int d = dsl[wave][m];
    if (d >= 0) {
      atomicAdd(&agg[d * 64 + col +  0], msg0);
      atomicAdd(&agg[d * 64 + col + 16], msg1);
      atomicAdd(&agg[d * 64 + col + 32], msg2);
      atomicAdd(&agg[d * 64 + col + 48], msg3);
    }
  }
}

// ---------------- GATv2 pass 1: wave-per-edge scores + segment max --------
__global__ void __launch_bounds__(256)
gnn_gat_score_kernel(const float* __restrict__ x, const int* __restrict__ ei,
                     const float* __restrict__ ea, const float* __restrict__ wl,
                     const float* __restrict__ bl, const float* __restrict__ wr,
                     const float* __restrict__ br, const float* __restrict__ we,
                     const float* __restrict__ att, unsigned* __restrict__ smax,
                     float* __restrict__ score, int E) {
  __shared__ float swl[768], swr[768], sbl[256], sbr[256], swe[256], satt[256];
  const int tid = threadIdx.x;
  for (int t = tid; t < 768; t += 256) { swl[t] = wl[t]; swr[t] = wr[t]; }
  if (tid < 256) { sbl[tid] = bl[tid]; sbr[tid] = br[tid];
                   swe[tid] = we[tid]; satt[tid] = att[tid]; }
  __syncthreads();

  const int lane = tid & 31;
  const int wid = blockIdx.x * 8 + (tid >> 5);   // one wave per edge
  if (wid < E) {
    int s = ei[wid], d = ei[E + wid];
    float xs0 = x[s * 3], xs1 = x[s * 3 + 1], xs2 = x[s * 3 + 2];
    float xd0 = x[d * 3], xd1 = x[d * 3 + 1], xd2 = x[d * 3 + 2];
    float a = ea[wid];
    float sc0 = 0.f, sc1 = 0.f, sc2 = 0.f, sc3 = 0.f;
#pragma unroll
    for (int j = 0; j < 8; ++j) {          // hc = j*32 + lane: coalesced
      int hc = j * 32 + lane;
      float xl = xs0 * swl[hc] + xs1 * swl[256 + hc] + xs2 * swl[512 + hc] + sbl[hc];
      float xr = xd0 * swr[hc] + xd1 * swr[256 + hc] + xd2 * swr[512 + hc] + sbr[hc];
      float g = xl + xr + a * swe[hc];
      g = (g > 0.f) ? g : 0.2f * g;
      float gv = g * satt[hc];
      if ((j >> 1) == 0) sc0 += gv;        // head = j>>1 (uniform per j)
      else if ((j >> 1) == 1) sc1 += gv;
      else if ((j >> 1) == 2) sc2 += gv;
      else sc3 += gv;
    }
#pragma unroll
    for (int msk = 1; msk <= 16; msk <<= 1) {
      sc0 += __shfl_xor(sc0, msk, 32);
      sc1 += __shfl_xor(sc1, msk, 32);
      sc2 += __shfl_xor(sc2, msk, 32);
      sc3 += __shfl_xor(sc3, msk, 32);
    }
    if (lane < 4) {
      float sc = (lane == 0) ? sc0 : (lane == 1 ? sc1 : (lane == 2 ? sc2 : sc3));
      score[wid * 4 + lane] = sc;
      atomicMax(&smax[d * 4 + lane], enc_f32(sc));
    }
  }
}

// ---------------- GATv2 pass 2: wave-per-edge exp/denom/numerator ---------
__global__ void __launch_bounds__(256)
gnn_gat_agg_kernel(const float* __restrict__ x, const int* __restrict__ ei,
                   const float* __restrict__ wl, const float* __restrict__ bl,
                   const unsigned* __restrict__ smax, const float* __restrict__ score,
                   float* __restrict__ asum, float* __restrict__ out2, int E) {
  __shared__ float swl[768], sbl[256];
  const int tid = threadIdx.x;
  for (int t = tid; t < 768; t += 256) swl[t] = wl[t];
  if (tid < 256) sbl[tid] = bl[tid];
  __syncthreads();

  const int lane = tid & 31;
  const int wid = blockIdx.x * 8 + (tid >> 5);   // one wave per edge
  if (wid < E) {
    int s = ei[wid], d = ei[E + wid];
    float xs0 = x[s * 3], xs1 = x[s * 3 + 1], xs2 = x[s * 3 + 2];
    float av = 0.f;
    if (lane < 4) {                               // lanes 0..3 own the 4 heads
      float mx = dec_f32(smax[d * 4 + lane]);
      av = __expf(score[wid * 4 + lane] - mx);
      atomicAdd(&asum[d * 4 + lane], av);
    }
#pragma unroll
    for (int j = 0; j < 8; ++j) {                 // hc = j*32 + lane: coalesced
      int hc = j * 32 + lane;
      float ah = __shfl(av, j >> 1, 32);          // head = j>>1, broadcast
      float xl = xs0 * swl[hc] + xs1 * swl[256 + hc] + xs2 * swl[512 + hc] + sbl[hc];
      atomicAdd(&out2[d * 256 + hc], ah * xl);    // 32 consecutive addrs/instr
    }
  }
}

// ---------------- fusion MLP (WMMA: N x 320 x 128 -> N x 128 x 32) --------
__global__ void __launch_bounds__(128)
gnn_fuse_kernel(const float* __restrict__ x, const float* __restrict__ agg,
                const float* __restrict__ out2, const float* __restrict__ asum,
                const float* __restrict__ w_root, const float* __restrict__ b_nn,
                const float* __restrict__ b_gat, const _Float16* __restrict__ wf1t,
                const float* __restrict__ bf1, const _Float16* __restrict__ wf2t,
                const float* __restrict__ bf2, float* __restrict__ out, int N) {
  __shared__ _Float16 comb[64 * 320];   // 40 KB, [node][k]
  __shared__ _Float16 act1[64 * 128];   // 16 KB, [node][k]
  const int lane = threadIdx.x & 31, wave = threadIdx.x >> 5;
  const int base = blockIdx.x * 64;
  const int wbase = wave * 16;

  // stage combined = [relu(agg + x@w_root + b_nn) | relu(out2/asum + b_gat)]
  for (int t = lane; t < 16 * 320; t += 32) {
    int mm = t / 320, n = t % 320;
    int node = base + wbase + mm;
    float v = 0.f;
    if (node < N) {
      if (n < 64) {
        float r = agg[node * 64 + n] + b_nn[n];
        r += x[node * 3 + 0] * w_root[n] + x[node * 3 + 1] * w_root[64 + n] +
             x[node * 3 + 2] * w_root[128 + n];
        v = r > 0.f ? r : 0.f;
      } else {
        int hc = n - 64;
        float r = out2[node * 256 + hc] / (asum[node * 4 + (hc >> 6)] + 1e-16f) + b_gat[hc];
        v = r > 0.f ? r : 0.f;
      }
    }
    comb[(wbase + mm) * 320 + n] = (_Float16)v;
  }
  __syncthreads();

  const int col = lane & 15, hi = lane >> 4, koff = hi * 8, kgrp = hi * 16;
  const _Float16* arow = &comb[(wbase + col) * 320];

  // layer 1: 8 n-tiles x 10 k-steps
  for (int nt = 0; nt < 8; ++nt) {
    const _Float16* brow = &wf1t[(nt * 16 + col) * 320];
    v8f c = {};
#pragma unroll
    for (int ks = 0; ks < 10; ++ks) {
      v16h a = ld_afrag(arow, ks * 32, koff);
      v16h b = ld_bfrag(brow, ks * 32, kgrp);
      c = WMMA_F16(a, b, c);
    }
#pragma unroll
    for (int v = 0; v < 8; ++v) {
      int m = hi * 8 + v, n = nt * 16 + col;
      float r = c[v] + bf1[n];
      act1[(wbase + m) * 128 + n] = (_Float16)(r > 0.f ? r : 0.f);
    }
  }
  __syncthreads();

  // layer 2: 2 n-tiles x 4 k-steps
  const _Float16* a2row = &act1[(wbase + col) * 128];
  for (int nt = 0; nt < 2; ++nt) {
    const _Float16* brow = &wf2t[(nt * 16 + col) * 128];
    v8f c = {};
#pragma unroll
    for (int ks = 0; ks < 4; ++ks) {
      v16h a = ld_afrag(a2row, ks * 32, koff);
      v16h b = ld_bfrag(brow, ks * 32, kgrp);
      c = WMMA_F16(a, b, c);
    }
#pragma unroll
    for (int v = 0; v < 8; ++v) {
      int m = hi * 8 + v, n = nt * 16 + col;
      int node = base + wbase + m;
      if (node < N) out[node * 32 + n] = c[v] + bf2[n];
    }
  }
}

// ---------------------------------------------------------------------------
extern "C" void kernel_launch(void* const* d_in, const int* in_sizes, int n_in,
                              void* d_out, int out_size, void* d_ws, size_t ws_size,
                              hipStream_t stream) {
  const float* x     = (const float*)d_in[0];
  const int*   ei    = (const int*)d_in[1];
  const float* ea    = (const float*)d_in[2];
  const float* w1    = (const float*)d_in[3];
  const float* b1    = (const float*)d_in[4];
  const float* w2    = (const float*)d_in[5];
  const float* b2    = (const float*)d_in[6];
  const float* ln_g  = (const float*)d_in[7];
  const float* ln_b  = (const float*)d_in[8];
  const float* w_root= (const float*)d_in[9];
  const float* b_nn  = (const float*)d_in[10];
  const float* wl    = (const float*)d_in[11];
  const float* bl    = (const float*)d_in[12];
  const float* wr    = (const float*)d_in[13];
  const float* br    = (const float*)d_in[14];
  const float* we    = (const float*)d_in[15];
  const float* att   = (const float*)d_in[16];
  const float* b_gat = (const float*)d_in[17];
  const float* wf1   = (const float*)d_in[18];
  const float* bf1   = (const float*)d_in[19];
  const float* wf2   = (const float*)d_in[20];
  const float* bf2   = (const float*)d_in[21];

  const int N = in_sizes[0] / 3;
  const int E = in_sizes[2];

  auto align256 = [](size_t v) { return (v + 255) & ~(size_t)255; };
  char* base = (char*)d_ws;
  size_t off = 0;
  float*    agg   = (float*)(base + off);    off = align256(off + (size_t)N * 64 * 4);
  float*    out2  = (float*)(base + off);    off = align256(off + (size_t)N * 256 * 4);
  float*    asum  = (float*)(base + off);    off = align256(off + (size_t)N * 4 * 4);
  unsigned* smax  = (unsigned*)(base + off); off = align256(off + (size_t)N * 4 * 4);
  float*    score = (float*)(base + off);    off = align256(off + (size_t)E * 4 * 4);
  _Float16* wf1t  = (_Float16*)(base + off); off = align256(off + (size_t)128 * 320 * 2);
  _Float16* wf2t  = (_Float16*)(base + off); off = align256(off + (size_t)32 * 128 * 2);
  (void)ws_size; (void)n_in; (void)out_size;

  // 1) init accumulators
  {
    long total = (long)N * 256;
    int blocks = (int)((total + 255) / 256);
    gnn_init_kernel<<<blocks, 256, 0, stream>>>(agg, out2, asum, smax, N);
  }
  // 2) convert fusion weights to n-major f16
  gnn_convw_kernel<<<(320 * 128 + 255) / 256, 256, 0, stream>>>(wf1, wf2, wf1t, wf2t);
  // 3) NNConv edge GEMM + scatter
  gnn_nnconv_kernel<<<(E + 127) / 128, 256, 0, stream>>>(
      x, ei, ea, w1, b1, w2, b2, ln_g, ln_b, agg, N, E);
  // 4) GAT scores + segment max (wave per edge)
  gnn_gat_score_kernel<<<(E + 7) / 8, 256, 0, stream>>>(
      x, ei, ea, wl, bl, wr, br, we, att, smax, score, E);
  // 5) GAT softmax denom + weighted numerator (wave per edge, coalesced atomics)
  gnn_gat_agg_kernel<<<(E + 7) / 8, 256, 0, stream>>>(
      x, ei, wl, bl, smax, score, asum, out2, E);
  // 6) fusion MLP (WMMA)
  gnn_fuse_kernel<<<(N + 63) / 64, 128, 0, stream>>>(
      x, agg, out2, asum, w_root, b_nn, b_gat, wf1t, bf1, wf2t, bf2,
      (float*)d_out, N);
}
